// Transformer_962072674378
// MI455X (gfx1250) — matrix operand order repaired and checked by reading, not compiled, not run on hardware
//
#include <hip/hip_runtime.h>
#include <hip/hip_bf16.h>
#include <math.h>

typedef __attribute__((ext_vector_type(16))) _Float16 v16h;
typedef __attribute__((ext_vector_type(8)))  _Float16 v8h;
typedef __attribute__((ext_vector_type(8)))  float    v8f;

#define QLEN   256
#define MLEN   256
#define KLENN  512
#define BSZ    8
#define DMODEL 1024
#define NHEAD  16
#define DHEAD  64
#define DINNER 4096
#define NTOKEN 32000
#define ROWS   (QLEN*BSZ)    // 2048
#define CATROWS (KLENN*BSZ)  // 4096
#define BH     (BSZ*NHEAD)   // 128

// ---------------- CDNA5 async global->LDS copy (ASYNCcnt path) ----------------
// Per-lane: copy 16 bytes from the lane's global address to the lane's LDS byte
// address. LDS byte address = low 32 bits of the flat address of a __shared__
// element (ISA 10.2: LDS aperture uses addr[31:0] as the LDS offset).
__device__ __forceinline__ void async_copy_b128(unsigned lds_off, const void* gaddr) {
  asm volatile("global_load_async_to_lds_b128 %0, %1, off"
               :: "v"(lds_off), "v"(gaddr) : "memory");
}
__device__ __forceinline__ void wait_async() {
  asm volatile("s_wait_asynccnt 0x0" ::: "memory");
}
__device__ __forceinline__ unsigned lds_addr(const void* p) {
  return (unsigned)(size_t)p;
}

// ---------------- block reductions (blockDim.x == 256) ----------------
__device__ __forceinline__ float block_reduce_sum(float v, float* sh) {
  int t = threadIdx.x;
  sh[t] = v; __syncthreads();
  for (int s = 128; s > 0; s >>= 1) { if (t < s) sh[t] += sh[t + s]; __syncthreads(); }
  float r = sh[0]; __syncthreads();
  return r;
}
__device__ __forceinline__ float block_reduce_max(float v, float* sh) {
  int t = threadIdx.x;
  sh[t] = v; __syncthreads();
  for (int s = 128; s > 0; s >>= 1) { if (t < s) sh[t] = fmaxf(sh[t], sh[t + s]); __syncthreads(); }
  float r = sh[0]; __syncthreads();
  return r;
}

// ---------------- WMMA fragment loaders (from LDS) ----------------
// A-fragment per ISA 16-bit A 16x32 layout: lane half hs: elems 0..7 = K hs*8..hs*8+7,
// elems 8..15 = K hs*8+16..hs*8+23.
__device__ __forceinline__ v16h load_frag_a(const _Float16* rowp, int hs) {
  v8h x = *(const v8h*)(rowp + hs * 8);
  v8h y = *(const v8h*)(rowp + hs * 8 + 16);
  v16h r;
#pragma unroll
  for (int i = 0; i < 8; i++) { r[i] = x[i]; r[i + 8] = y[i]; }
  return r;
}
// B-fragment (same K order as A): lane half hs: elems 0..7 = K hs*8..hs*8+7, elems 8..15 = K hs*8+16..hs*8+23.
__device__ __forceinline__ v16h load_frag_b(const _Float16* rowp, int hs) {
  v8h x = *(const v8h*)(rowp + hs * 8);
  v8h y = *(const v8h*)(rowp + hs * 8 + 16);
  v16h r;
#pragma unroll
  for (int i = 0; i < 8; i++) { r[i] = x[i]; r[i + 8] = y[i]; }
  return r;
}

// ---------------- generic batched f16 WMMA GEMM ----------------
// C[M,N] = A[M,K] * B  (+bias[n]) (+relu).  C is f32 (OUT16=0) or f16 (OUT16=1).
// A row-major lda=K.  TB=0: B row-major [K,N].  TB=1: B row-major [N,K] (i.e. B^T).
// Block tile 128x64, 8 waves as 4x2, wave tile 32x32 (2x2 WMMA 16x16x32 f16->f32).
// A tile (and B tile when TB) staged via GLOBAL_LOAD_ASYNC_TO_LDS_B128.
template <bool TB, int ACT, bool OUT16>
__global__ __launch_bounds__(256)
void gemm_wmma(const _Float16* __restrict__ A, const _Float16* __restrict__ B,
               const float* __restrict__ bias, void* __restrict__ Cv,
               int M, int N, int K,
               long long strideA, long long strideB, long long strideC) {
  (void)M;
  A += (long long)blockIdx.z * strideA;
  B += (long long)blockIdx.z * strideB;
  const int m0 = blockIdx.y * 128;
  const int n0 = blockIdx.x * 64;

  __shared__ _Float16 As[128][32];
  __shared__ _Float16 Bs[64][32];

  const int tid = threadIdx.x;
  const int lane = tid & 31;
  const int w = tid >> 5;
  const int wr = w >> 1;   // 0..3  (M sub-tile)
  const int wc = w & 1;    // 0..1  (N sub-tile)
  const int r = lane & 15;
  const int hs = lane >> 4;

  v8f acc[2][2] = {};

  // per-thread staging coordinates (fixed across K loop)
  const int arow = tid >> 1, ac0 = (tid & 1) * 16;
  const unsigned a_lds0 = lds_addr(&As[arow][ac0]);
  const unsigned a_lds1 = lds_addr(&As[arow][ac0 + 8]);

  for (int kk = 0; kk < K; kk += 32) {
    // ---- stage A tile 128x32 via async copy (2 x 16B per thread) ----
    {
      const _Float16* src = A + (size_t)(m0 + arow) * K + kk + ac0;
      async_copy_b128(a_lds0, src);
      async_copy_b128(a_lds1, src + 8);
    }
    // ---- stage B tile into Bs[n][k] ----
    if constexpr (TB) {
      int n = tid >> 2, k0 = (tid & 3) * 8;
      const _Float16* src = B + (size_t)(n0 + n) * K + kk + k0;
      async_copy_b128(lds_addr(&Bs[n][k0]), src);
    } else {
      int k = tid >> 3, ng = (tid & 7) * 8;
      const _Float16* src = B + (size_t)(kk + k) * N + n0 + ng;
      v8h tmp = *(const v8h*)src;
      __builtin_prefetch(src + (size_t)32 * N, 0, 1);  // speculative: safe past end
#pragma unroll
      for (int e = 0; e < 8; e++) Bs[ng + e][k] = tmp[e];
    }
    wait_async();
    __syncthreads();

    v16h af[2], bf[2];
#pragma unroll
    for (int mt = 0; mt < 2; mt++) af[mt] = load_frag_a(&As[wr * 32 + mt * 16 + r][0], hs);
#pragma unroll
    for (int nt = 0; nt < 2; nt++) bf[nt] = load_frag_b(&Bs[wc * 32 + nt * 16 + r][0], hs);

#pragma unroll
    for (int mt = 0; mt < 2; mt++)
#pragma unroll
      for (int nt = 0; nt < 2; nt++)
        acc[mt][nt] = __builtin_amdgcn_wmma_f32_16x16x32_f16(
            false, af[mt], false, bf[nt], (short)0, acc[mt][nt], false, false);
    __syncthreads();
  }

  // ---- epilogue: C 16x16 layout: col = (lane&15), rows = hs*8 + j ----
  float*    Cf = (float*)Cv + (OUT16 ? 0 : (long long)blockIdx.z * strideC);
  _Float16* Ch = (_Float16*)Cv + (OUT16 ? (long long)blockIdx.z * strideC : 0);
#pragma unroll
  for (int mt = 0; mt < 2; mt++) {
#pragma unroll
    for (int nt = 0; nt < 2; nt++) {
      int col = n0 + wc * 32 + nt * 16 + r;
      int rowb = m0 + wr * 32 + mt * 16 + hs * 8;
      float bv = bias ? bias[col] : 0.0f;
#pragma unroll
      for (int j = 0; j < 8; j++) {
        float v = acc[mt][nt][j] + bv;
        if (ACT == 1) v = v > 0.0f ? v : 0.0f;
        size_t cidx = (size_t)(rowb + j) * N + col;
        if constexpr (OUT16) Ch[cidx] = (_Float16)v;
        else                 Cf[cidx] = v;
      }
    }
  }
}

// ---------------- elementwise / small kernels ----------------
__global__ void embed_kernel(const int* __restrict__ tok, const float* __restrict__ emb,
                             float* __restrict__ h) {
  size_t idx = (size_t)blockIdx.x * 256 + threadIdx.x;  // < ROWS*DMODEL
  size_t rb = idx >> 10; int d = idx & 1023;
  h[idx] = emb[(size_t)tok[rb] * DMODEL + d] * 32.0f;   // sqrt(1024)
}

__global__ void posemb_kernel(_Float16* __restrict__ pe) {
  size_t idx = (size_t)blockIdx.x * 256 + threadIdx.x;  // < KLENN*DMODEL
  int j = (int)(idx >> 10), d = (int)(idx & 1023);
  float pos = (float)(KLENN - 1 - j);
  int d2 = (d < 512) ? d : d - 512;
  float invf = __expf(-((float)(2 * d2) / (float)DMODEL) * 9.210340371976184f); // ln(1e4)
  float a = pos * invf;
  pe[idx] = (_Float16)((d < 512) ? sinf(a) : cosf(a));
}

__global__ void build_cat_kernel(const float* __restrict__ mems_l, const float* __restrict__ h,
                                 _Float16* __restrict__ cat) {
  size_t idx = (size_t)blockIdx.x * 256 + threadIdx.x;  // < CATROWS*DMODEL
  size_t rb = idx >> 10;
  int j = (int)(rb >> 3);
  float v = (j < MLEN) ? mems_l[idx] : h[idx - (size_t)MLEN * BSZ * DMODEL];
  cat[idx] = (_Float16)v;
}

__global__ void cvt_f16_kernel(const float* __restrict__ in, _Float16* __restrict__ out, size_t n) {
  size_t i = (size_t)blockIdx.x * 256 + threadIdx.x;
  if (i < n) out[i] = (_Float16)in[i];
}

__global__ void split_qkv_kernel(const float* __restrict__ wh,
                                 const float* __restrict__ rwb, const float* __restrict__ rrb,
                                 _Float16* __restrict__ qrw, _Float16* __restrict__ qrr,
                                 _Float16* __restrict__ kk, _Float16* __restrict__ vv) {
  size_t idx = (size_t)blockIdx.x * 256 + threadIdx.x;  // < KLENN*BSZ*DMODEL
  int d = (int)(idx & 1023);
  size_t rb = idx >> 10;
  int b = (int)(rb & 7);
  int j = (int)(rb >> 3);
  int hh = d >> 6, kd = d & 63;
  const float* src = wh + rb * 3072;
  size_t kvidx = (((size_t)(b * NHEAD + hh) * KLENN) + j) * DHEAD + kd;
  kk[kvidx] = (_Float16)src[1024 + d];
  vv[kvidx] = (_Float16)src[2048 + d];
  if (j >= MLEN) {
    int i = j - MLEN;
    float q = src[d];
    size_t qidx = (((size_t)(b * NHEAD + hh) * QLEN) + i) * DHEAD + kd;
    qrw[qidx] = (_Float16)(q + rwb[d]);
    qrr[qidx] = (_Float16)(q + rrb[d]);
  }
}

__global__ void rk_expand_kernel(const float* __restrict__ rk, _Float16* __restrict__ out) {
  size_t idx = (size_t)blockIdx.x * 256 + threadIdx.x;  // < BH*KLENN*DHEAD
  int kd = (int)(idx & 63);
  size_t t = idx >> 6;
  int j = (int)(t & 511);
  int hh = (int)((t >> 9) & 15);
  out[idx] = (_Float16)rk[(size_t)j * DMODEL + hh * DHEAD + kd];
}

// score = (AC[i,j] + BDraw[i, j+255-i]) * 1/8, masked (j > i+MLEN), softmax over j
__global__ __launch_bounds__(256)
void attn_softmax_kernel(const float* __restrict__ AC, const float* __restrict__ BD,
                         _Float16* __restrict__ prob) {
  __shared__ float sh[256];
  int i = blockIdx.x;    // 0..QLEN-1
  int bh = blockIdx.y;   // 0..BH-1
  const float* ac = AC + ((size_t)bh * QLEN + i) * KLENN;
  const float* bd = BD + ((size_t)bh * QLEN + i) * KLENN;
  _Float16* pr = prob + ((size_t)bh * QLEN + i) * KLENN;
  int t = threadIdx.x;
  float s[2]; bool valid[2];
#pragma unroll
  for (int u = 0; u < 2; u++) {
    int j = t + u * 256;
    valid[u] = (j <= i + MLEN);
    int js = j + (MLEN - 1) - i;   // rel_shift index; in-range whenever valid
    s[u] = valid[u] ? (ac[j] + bd[js]) * 0.125f : -3.0e38f;
  }
  float m = block_reduce_max(fmaxf(s[0], s[1]), sh);
  float e0 = valid[0] ? __expf(s[0] - m) : 0.0f;
  float e1 = valid[1] ? __expf(s[1] - m) : 0.0f;
  float tot = block_reduce_sum(e0 + e1, sh);
  float inv = 1.0f / tot;
  pr[t]       = (_Float16)(e0 * inv);
  pr[t + 256] = (_Float16)(e1 * inv);
}

__global__ void vec_rearrange_kernel(const float* __restrict__ ao, _Float16* __restrict__ vec) {
  size_t idx = (size_t)blockIdx.x * 256 + threadIdx.x;  // < BH*QLEN*DHEAD
  int kd = (int)(idx & 63);
  size_t t = idx >> 6;
  int i = (int)(t & 255);
  int bh = (int)(t >> 8);
  int hh = bh & 15, b = bh >> 4;
  vec[((size_t)(i * BSZ + b)) * DMODEL + hh * DHEAD + kd] = (_Float16)ao[idx];
}

__global__ __launch_bounds__(256)
void add_ln_kernel(const float* __restrict__ x, float* __restrict__ h,
                   const float* __restrict__ g, const float* __restrict__ b,
                   _Float16* __restrict__ h16) {
  __shared__ float sh[256];
  size_t row = blockIdx.x;
  const float* xp = x + row * DMODEL;
  float* hp = h + row * DMODEL;
  int t = threadIdx.x;
  float v[4]; float lsum = 0.f, lsq = 0.f;
#pragma unroll
  for (int u = 0; u < 4; u++) {
    int d = t + u * 256;
    v[u] = xp[d] + hp[d];
    lsum += v[u]; lsq += v[u] * v[u];
  }
  float tot = block_reduce_sum(lsum, sh);
  float tot2 = block_reduce_sum(lsq, sh);
  float mean = tot * (1.0f / DMODEL);
  float var = tot2 * (1.0f / DMODEL) - mean * mean;
  float inv = rsqrtf(var + 1.0e-3f);
#pragma unroll
  for (int u = 0; u < 4; u++) {
    int d = t + u * 256;
    float y = (v[u] - mean) * inv * g[d] + b[d];
    hp[d] = y;
    h16[row * DMODEL + d] = (_Float16)y;
  }
}

__global__ __launch_bounds__(256)
void nll_kernel(const float* __restrict__ logits, const int* __restrict__ target,
                const float* __restrict__ vm, float* __restrict__ nll) {
  __shared__ float sh[256];
  size_t row = blockIdx.x;
  const float* lg = logits + row * (size_t)NTOKEN;
  int t = threadIdx.x;
  float m = -3.0e38f;
  for (int j = t; j < NTOKEN; j += 256) m = fmaxf(m, lg[j]);
  m = block_reduce_max(m, sh);
  float s = 0.0f;
  for (int j = t; j < NTOKEN; j += 256) s += __expf(lg[j] - m);
  s = block_reduce_sum(s, sh);
  if (t == 0) {
    int tg = target[row];
    nll[row] = (logf(s) + m - lg[tg]) * vm[row];
  }
}

__global__ __launch_bounds__(256)
void mean_kernel(const float* __restrict__ nll, float* __restrict__ out) {
  __shared__ float sh[256];
  int t = threadIdx.x;
  float lsum = 0.f;
#pragma unroll
  for (int u = 0; u < 8; u++) lsum += nll[t + u * 256];
  float tot = block_reduce_sum(lsum, sh);
  if (t == 0) out[0] = tot / (float)ROWS;
}

// ---------------- host side ----------------
static void launch_gemm(int tb, int act, int out16,
                        const _Float16* A, const _Float16* B,
                        const float* bias, void* C, int M, int N, int K,
                        long long sA, long long sB, long long sC, int batch,
                        hipStream_t stream) {
  dim3 grid(N / 64, M / 128, batch), blk(256);
  if (tb) {
    if (out16) {
      if (act) gemm_wmma<true, 1, true ><<<grid, blk, 0, stream>>>(A, B, bias, C, M, N, K, sA, sB, sC);
      else     gemm_wmma<true, 0, true ><<<grid, blk, 0, stream>>>(A, B, bias, C, M, N, K, sA, sB, sC);
    } else {
      if (act) gemm_wmma<true, 1, false><<<grid, blk, 0, stream>>>(A, B, bias, C, M, N, K, sA, sB, sC);
      else     gemm_wmma<true, 0, false><<<grid, blk, 0, stream>>>(A, B, bias, C, M, N, K, sA, sB, sC);
    }
  } else {
    if (out16) {
      if (act) gemm_wmma<false, 1, true ><<<grid, blk, 0, stream>>>(A, B, bias, C, M, N, K, sA, sB, sC);
      else     gemm_wmma<false, 0, true ><<<grid, blk, 0, stream>>>(A, B, bias, C, M, N, K, sA, sB, sC);
    } else {
      if (act) gemm_wmma<false, 1, false><<<grid, blk, 0, stream>>>(A, B, bias, C, M, N, K, sA, sB, sC);
      else     gemm_wmma<false, 0, false><<<grid, blk, 0, stream>>>(A, B, bias, C, M, N, K, sA, sB, sC);
    }
  }
}

static void launch_cvt(const float* in, _Float16* out, size_t n, hipStream_t stream) {
  unsigned int blocks = (unsigned int)((n + 255) / 256);
  cvt_f16_kernel<<<blocks, 256, 0, stream>>>(in, out, n);
}

extern "C" void kernel_launch(void* const* d_in, const int* in_sizes, int n_in,
                              void* d_out, int out_size, void* d_ws, size_t ws_size,
                              hipStream_t stream) {
  (void)in_sizes; (void)n_in; (void)out_size; (void)ws_size;
  const int*   dec_inp    = (const int*)d_in[0];
  const int*   target     = (const int*)d_in[1];
  const float* valid_mask = (const float*)d_in[2];
  const float* mems       = (const float*)d_in[3];
  const float* vocab_emb  = (const float*)d_in[4];
  const float* out_w      = (const float*)d_in[5];
  const float* out_b      = (const float*)d_in[6];
  const float* qkv_w      = (const float*)d_in[7];
  const float* r_w        = (const float*)d_in[8];
  const float* o_w        = (const float*)d_in[9];
  const float* ln1_g      = (const float*)d_in[10];
  const float* ln1_b      = (const float*)d_in[11];
  const float* ff_w1      = (const float*)d_in[12];
  const float* ff_b1      = (const float*)d_in[13];
  const float* ff_w2      = (const float*)d_in[14];
  const float* ff_b2      = (const float*)d_in[15];
  const float* ln2_g      = (const float*)d_in[16];
  const float* ln2_b      = (const float*)d_in[17];
  const float* r_w_bias   = (const float*)d_in[18];
  const float* r_r_bias   = (const float*)d_in[19];

  unsigned char* base = (unsigned char*)d_ws;
  size_t off = 0;
  auto alloc = [&](size_t bytes) -> void* {
    void* p = base + off;
    off = (off + bytes + 255) & ~(size_t)255;
    return p;
  };

  float*    h32     = (float*)   alloc((size_t)ROWS * DMODEL * 4);
  _Float16* h16     = (_Float16*)alloc((size_t)ROWS * DMODEL * 2);
  _Float16* cat16   = (_Float16*)alloc((size_t)CATROWS * DMODEL * 2);
  _Float16* pos16   = (_Float16*)alloc((size_t)KLENN * DMODEL * 2);
  _Float16* wbuf    = (_Float16*)alloc((size_t)NTOKEN * DMODEL * 2);
  float*    wheads  = (float*)   alloc((size_t)CATROWS * 3072 * 4);
  _Float16* qrw16   = (_Float16*)alloc((size_t)BH * QLEN * DHEAD * 2);
  _Float16* qrr16   = (_Float16*)alloc((size_t)BH * QLEN * DHEAD * 2);
  _Float16* k16     = (_Float16*)alloc((size_t)BH * KLENN * DHEAD * 2);
  _Float16* v16     = (_Float16*)alloc((size_t)BH * KLENN * DHEAD * 2);
  float*    rk32    = (float*)   alloc((size_t)KLENN * DMODEL * 4);
  _Float16* rk16    = (_Float16*)alloc((size_t)BH * KLENN * DHEAD * 2);
  float*    AC32    = (float*)   alloc((size_t)BH * QLEN * KLENN * 4);
  float*    BD32    = (float*)   alloc((size_t)BH * QLEN * KLENN * 4);
  _Float16* prob16  = (_Float16*)alloc((size_t)BH * QLEN * KLENN * 2);
  float*    ao32    = (float*)   alloc((size_t)BH * QLEN * DHEAD * 4);
  _Float16* vec16   = (_Float16*)alloc((size_t)ROWS * DMODEL * 2);
  float*    proj32  = (float*)   alloc((size_t)ROWS * DMODEL * 4);
  _Float16* ff1_16  = (_Float16*)alloc((size_t)ROWS * DINNER * 2);
  float*    ff2_32  = (float*)   alloc((size_t)ROWS * DMODEL * 4);
  float*    logits  = (float*)   alloc((size_t)ROWS * NTOKEN * 4);
  float*    nllbuf  = (float*)   alloc((size_t)ROWS * 4);

  embed_kernel<<<ROWS * DMODEL / 256, 256, 0, stream>>>(dec_inp, vocab_emb, h32);
  posemb_kernel<<<KLENN * DMODEL / 256, 256, 0, stream>>>(pos16);

  for (int l = 0; l < 2; l++) {
    const float* mems_l = mems + (size_t)l * MLEN * BSZ * DMODEL;
    build_cat_kernel<<<CATROWS * DMODEL / 256, 256, 0, stream>>>(mems_l, h32, cat16);

    // qkv projection: [4096,1024] x [1024,3072]
    launch_cvt(qkv_w + (size_t)l * DMODEL * 3072, wbuf, (size_t)DMODEL * 3072, stream);
    launch_gemm(0, 0, 0, cat16, wbuf, nullptr, wheads, CATROWS, 3072, DMODEL, 0, 0, 0, 1, stream);

    // r_head_k: [512,1024] x [1024,1024]
    launch_cvt(r_w + (size_t)l * DMODEL * DMODEL, wbuf, (size_t)DMODEL * DMODEL, stream);
    launch_gemm(0, 0, 0, pos16, wbuf, nullptr, rk32, KLENN, DMODEL, DMODEL, 0, 0, 0, 1, stream);

    split_qkv_kernel<<<KLENN * BSZ * DMODEL / 256, 256, 0, stream>>>(
        wheads, r_w_bias, r_r_bias, qrw16, qrr16, k16, v16);
    rk_expand_kernel<<<BH * KLENN * DHEAD / 256, 256, 0, stream>>>(rk32, rk16);

    // AC = (q+rwb) k^T ; BD = (q+rrb) rk^T  — batched over (b,h)
    launch_gemm(1, 0, 0, qrw16, k16, nullptr, AC32, QLEN, KLENN, DHEAD,
                (long long)QLEN * DHEAD, (long long)KLENN * DHEAD, (long long)QLEN * KLENN, BH, stream);
    launch_gemm(1, 0, 0, qrr16, rk16, nullptr, BD32, QLEN, KLENN, DHEAD,
                (long long)QLEN * DHEAD, (long long)KLENN * DHEAD, (long long)QLEN * KLENN, BH, stream);

    attn_softmax_kernel<<<dim3(QLEN, BH), 256, 0, stream>>>(AC32, BD32, prob16);

    // out = prob @ V — batched
    launch_gemm(0, 0, 0, prob16, v16, nullptr, ao32, QLEN, DHEAD, KLENN,
                (long long)QLEN * KLENN, (long long)KLENN * DHEAD, (long long)QLEN * DHEAD, BH, stream);
    vec_rearrange_kernel<<<BH * QLEN * DHEAD / 256, 256, 0, stream>>>(ao32, vec16);

    // o projection + residual LN
    launch_cvt(o_w + (size_t)l * DMODEL * DMODEL, wbuf, (size_t)DMODEL * DMODEL, stream);
    launch_gemm(0, 0, 0, vec16, wbuf, nullptr, proj32, ROWS, DMODEL, DMODEL, 0, 0, 0, 1, stream);
    add_ln_kernel<<<ROWS, 256, 0, stream>>>(proj32, h32, ln1_g + l * DMODEL, ln1_b + l * DMODEL, h16);

    // FF: relu(h w1 + b1) -> f16 directly, relu(. w2 + b2), residual LN
    launch_cvt(ff_w1 + (size_t)l * DMODEL * DINNER, wbuf, (size_t)DMODEL * DINNER, stream);
    launch_gemm(0, 1, 1, h16, wbuf, ff_b1 + l * DINNER, ff1_16, ROWS, DINNER, DMODEL, 0, 0, 0, 1, stream);
    launch_cvt(ff_w2 + (size_t)l * DINNER * DMODEL, wbuf, (size_t)DINNER * DMODEL, stream);
    launch_gemm(0, 1, 0, ff1_16, wbuf, ff_b2 + l * DMODEL, ff2_32, ROWS, DMODEL, DINNER, 0, 0, 0, 1, stream);
    add_ln_kernel<<<ROWS, 256, 0, stream>>>(ff2_32, h32, ln2_g + l * DMODEL, ln2_b + l * DMODEL, h16);
  }

  // logits = h @ out_w^T + out_b : [2048,1024] x [32000,1024]^T
  launch_cvt(out_w, wbuf, (size_t)NTOKEN * DMODEL, stream);
  launch_gemm(1, 0, 0, h16, wbuf, out_b, logits, ROWS, NTOKEN, DMODEL, 0, 0, 0, 1, stream);

  nll_kernel<<<ROWS, 256, 0, stream>>>(logits, target, valid_mask, nllbuf);
  mean_kernel<<<1, 256, 0, stream>>>(nllbuf, (float*)d_out);
}
